// MultiHeadAttention_81561428951442
// MI455X (gfx1250) — matrix operand order repaired and checked by reading, not compiled, not run on hardware
//
#include <hip/hip_runtime.h>
#include <hip/hip_bf16.h>

#define D_MODEL   1024
#define NUM_HEADS 16
#define HEAD_DIM  64
#define BATCH     2
#define SEQ       2048
#define MTOK      (BATCH * SEQ)   // 4096 tokens

typedef __bf16 bf16;
typedef bf16  v16bf __attribute__((ext_vector_type(16)));
typedef bf16  v8bf  __attribute__((ext_vector_type(8)));
typedef float v8f   __attribute__((ext_vector_type(8)));
typedef float v4f   __attribute__((ext_vector_type(4)));
typedef int   v4i   __attribute__((ext_vector_type(4)));

#if __has_builtin(__builtin_amdgcn_global_load_async_to_lds_b128) && \
    __has_builtin(__builtin_amdgcn_s_wait_asynccnt)
#define USE_ASYNC_LDS 1
#endif

#ifdef USE_ASYNC_LDS
// Builtin signature (from hipcc diagnostic): (v4i AS1*, v4i AS3*, imm int, imm int)
typedef __attribute__((address_space(1))) v4i as1_v4i;
typedef __attribute__((address_space(3))) v4i as3_v4i;
#endif

union BF16Frag { v16bf v; v8bf h[2]; };

__device__ __forceinline__ v8f wmma_bf16(const BF16Frag& a, const BF16Frag& b, v8f c) {
  // D = A(16x32 bf16) x B(32x16 bf16) + C(16x16 f32)
  return __builtin_amdgcn_wmma_f32_16x16x32_bf16(false, a.v, false, b.v,
                                                 (short)0, c, false, false);
}

// ---------------------------------------------------------------------------
// GEMM: out[m,n] = sum_k A[m,k] * W[n,k] + bias[n]   (M=4096, N=K=1024)
// MODE 0: A fp32, out bf16 head-split [B,H,S,hd]
// MODE 2: A fp32, out bf16 head-split transposed [B,H,hd,S]   (for V)
// MODE 1: A bf16, out fp32 [M, D_MODEL]                       (final proj)
// Block tile 128x128x32, 256 threads = 8 waves, wave tile 32x64.
// Register-prefetch pipeline: next K-tile's global loads overlap the WMMAs.
// ---------------------------------------------------------------------------
template <int MODE>
__global__ __launch_bounds__(256) void gemm_bias_kernel(
    const void* __restrict__ Ap, const float* __restrict__ W,
    const float* __restrict__ bias, void* __restrict__ outp)
{
  constexpr int LD = 40;                       // LDS row stride (bf16), 16B-aligned chunks
  __shared__ __align__(16) bf16 ldsA[128 * LD];
  __shared__ __align__(16) bf16 ldsB[128 * LD];

  const int tid  = threadIdx.x;
  const int lane = tid & 31;
  const int wave = tid >> 5;
  const int wm   = (wave >> 1) * 32;           // wave row offset in block tile
  const int wn   = (wave & 1) * 64;            // wave col offset in block tile
  const int m0   = blockIdx.y * 128;
  const int n0   = blockIdx.x * 128;

  const int lrow = tid >> 1;                   // 0..127 : tile row this thread loads
  const int lcol = (tid & 1) * 16;             // 0 or 16: half-row of 16 elements

  v8f acc[2][4];
  #pragma unroll
  for (int i = 0; i < 2; ++i)
    #pragma unroll
    for (int j = 0; j < 4; ++j)
      #pragma unroll
      for (int e = 0; e < 8; ++e) acc[i][j][e] = 0.0f;

  const float* gA  = (const float*)Ap + (size_t)(m0 + lrow) * D_MODEL + lcol;
  const bf16*  gAb = (const bf16*) Ap + (size_t)(m0 + lrow) * D_MODEL + lcol;
  const float* gW  = W + (size_t)(n0 + lrow) * D_MODEL + lcol;

  v4f  rA[4], rB[4];                           // raw fp32 prefetch registers
  v8bf rAb[2];                                 // bf16 prefetch (MODE 1)

  auto prefetch = [&](int K0) {
    if (MODE == 1) {
      rAb[0] = *(const v8bf*)(gAb + K0);
      rAb[1] = *(const v8bf*)(gAb + K0 + 8);
    } else {
      #pragma unroll
      for (int q = 0; q < 4; ++q) rA[q] = *(const v4f*)(gA + K0 + q * 4);
    }
    #pragma unroll
    for (int q = 0; q < 4; ++q) rB[q] = *(const v4f*)(gW + K0 + q * 4);
  };
  auto commit = [&]() {                        // convert + store staged tile to LDS
    v8bf lo, hi;
    if (MODE == 1) { lo = rAb[0]; hi = rAb[1]; }
    else {
      #pragma unroll
      for (int e = 0; e < 8; ++e) {
        lo[e] = (bf16)rA[e >> 2][e & 3];
        hi[e] = (bf16)rA[2 + (e >> 2)][e & 3];
      }
    }
    *(v8bf*)&ldsA[lrow * LD + lcol]     = lo;
    *(v8bf*)&ldsA[lrow * LD + lcol + 8] = hi;
    #pragma unroll
    for (int e = 0; e < 8; ++e) {
      lo[e] = (bf16)rB[e >> 2][e & 3];
      hi[e] = (bf16)rB[2 + (e >> 2)][e & 3];
    }
    *(v8bf*)&ldsB[lrow * LD + lcol]     = lo;
    *(v8bf*)&ldsB[lrow * LD + lcol + 8] = hi;
  };

  prefetch(0);
  for (int k0 = 0; k0 < D_MODEL; k0 += 32) {
    commit();
    __syncthreads();
    if (k0 + 32 < D_MODEL) prefetch(k0 + 32);  // overlaps the WMMAs below

    const int ar = lane & 15;
    const int ak = (lane >> 4) * 8;            // A: K chunks {ak, ak+16}
    const int bk = (lane >> 4) * 8;            // B: K chunks {bk, bk+16}
    BF16Frag afr[2], bfr[4];
    #pragma unroll
    for (int i = 0; i < 2; ++i) {
      const bf16* p = &ldsA[(wm + i * 16 + ar) * LD];
      afr[i].h[0] = *(const v8bf*)(p + ak);
      afr[i].h[1] = *(const v8bf*)(p + ak + 16);
    }
    #pragma unroll
    for (int j = 0; j < 4; ++j) {
      const bf16* p = &ldsB[(wn + j * 16 + ar) * LD];
      bfr[j].h[0] = *(const v8bf*)(p + bk);
      bfr[j].h[1] = *(const v8bf*)(p + bk + 16);
    }
    #pragma unroll
    for (int i = 0; i < 2; ++i)
      #pragma unroll
      for (int j = 0; j < 4; ++j)
        acc[i][j] = wmma_bf16(afr[i], bfr[j], acc[i][j]);
    __syncthreads();                           // LDS free for next commit
  }

  // ---- epilogue: bias + scatter (C/D layout: M = v + 8*(lane>=16), N = lane&15) ----
  const int mlane = (lane >> 4) * 8;
  const int nlane = lane & 15;
  #pragma unroll
  for (int i = 0; i < 2; ++i) {
    #pragma unroll
    for (int j = 0; j < 4; ++j) {
      #pragma unroll
      for (int v = 0; v < 8; ++v) {
        const int m = m0 + wm + i * 16 + mlane + v;
        const int n = n0 + wn + j * 16 + nlane;
        const float val = acc[i][j][v] + bias[n];
        if (MODE == 1) {
          ((float*)outp)[(size_t)m * D_MODEL + n] = val;
        } else {
          const int bb = m >> 11;              // m / SEQ
          const int s  = m & (SEQ - 1);
          const int hh = n >> 6;
          const int d  = n & (HEAD_DIM - 1);
          const int bh = bb * NUM_HEADS + hh;
          if (MODE == 0)
            ((bf16*)outp)[((size_t)bh * SEQ + s) * HEAD_DIM + d] = (bf16)val;
          else
            ((bf16*)outp)[((size_t)bh * HEAD_DIM + d) * SEQ + s] = (bf16)val;
        }
      }
    }
  }
}

// ---------------------------------------------------------------------------
// Flash-style attention. Block = 128 query rows of one (b,h); 8 waves x 16 rows.
// Q [B,H,S,64] bf16, K [B,H,S,64] bf16, Vt [B,H,64,S] bf16,
// out = merged context [B,S,D_MODEL] bf16.
// K/V tiles double-buffered via GLOBAL_LOAD_ASYNC_TO_LDS_B128 when available.
// ---------------------------------------------------------------------------
__global__ __launch_bounds__(256) void attention_kernel(
    const bf16* __restrict__ Qh, const bf16* __restrict__ Kh,
    const bf16* __restrict__ Vt, const unsigned char* __restrict__ mask,
    bf16* __restrict__ out)
{
  constexpr int LDK  = 72;                     // LDS row stride, 16B-aligned chunks
  constexpr int NIT  = SEQ / 64;
#ifdef USE_ASYNC_LDS
  constexpr int NBUF = 2;
#else
  constexpr int NBUF = 1;
#endif
  __shared__ __align__(16) bf16 ldsK[NBUF][64 * LDK];
  __shared__ __align__(16) bf16 ldsV[NBUF][64 * LDK];
  __shared__ __align__(16) bf16 ldsP[8 * 16 * LDK];

  const int tid  = threadIdx.x;
  const int lane = tid & 31;
  const int wave = tid >> 5;
  const int bh   = blockIdx.y;                 // b*16 + h
  const int b    = bh >> 4;
  const int h    = bh & 15;
  const int q0   = blockIdx.x * 128 + wave * 16;

  // Q A-fragments (two K-steps of 32), pinned in registers for the whole kernel.
  BF16Frag qf[2];
  {
    const int qr = q0 + (lane & 15);
    const int aq = (lane >> 4) * 8;
    const bf16* qp = Qh + ((size_t)bh * SEQ + qr) * HEAD_DIM;
    #pragma unroll
    for (int s = 0; s < 2; ++s) {
      qf[s].h[0] = *(const v8bf*)(qp + s * 32 + aq);
      qf[s].h[1] = *(const v8bf*)(qp + s * 32 + aq + 16);
    }
  }

  v8f   o[4];
  float mrow[8], lrow[8];
  #pragma unroll
  for (int d = 0; d < 4; ++d)
    #pragma unroll
    for (int v = 0; v < 8; ++v) o[d][v] = 0.0f;
  #pragma unroll
  for (int v = 0; v < 8; ++v) { mrow[v] = -3.0e38f; lrow[v] = 0.0f; }

  const int   krow  = tid >> 2;                // 0..63 : staged tile row
  const int   kcol  = (tid & 3) * 16;
  const float scale = 0.125f;                  // 1/sqrt(64)
  bf16* pw = &ldsP[wave * 16 * LDK];

  const int nr   = lane & 15;
  const int kb16 = (lane >> 4) * 8;
  const int ak   = (lane >> 4) * 8;

  // -------- per-tile compute: scores -> online softmax -> O += P x V --------
  auto compute_tile = [&](int kb, const bf16* lk, const bf16* lv) {
    float sc[4][8];
    #pragma unroll
    for (int tn = 0; tn < 4; ++tn) {
      v8f s;
      #pragma unroll
      for (int v = 0; v < 8; ++v) s[v] = 0.0f;
      #pragma unroll
      for (int ks = 0; ks < 2; ++ks) {
        BF16Frag kf;
        const bf16* p = &lk[(tn * 16 + nr) * LDK + ks * 32 + kb16];
        kf.h[0] = *(const v8bf*)p;
        kf.h[1] = *(const v8bf*)(p + 16);
        s = wmma_bf16(qf[ks], kf, s);
      }
      const unsigned char mk = mask[(size_t)b * SEQ + kb + tn * 16 + nr];
      #pragma unroll
      for (int v = 0; v < 8; ++v) sc[tn][v] = mk ? s[v] * scale : -1.0e9f;
    }

    float tm[8];
    #pragma unroll
    for (int v = 0; v < 8; ++v)
      tm[v] = fmaxf(fmaxf(sc[0][v], sc[1][v]), fmaxf(sc[2][v], sc[3][v]));
    #pragma unroll
    for (int off = 8; off >= 1; off >>= 1)
      #pragma unroll
      for (int v = 0; v < 8; ++v)
        tm[v] = fmaxf(tm[v], __shfl_xor(tm[v], off, 32));

    float corr[8], rs[8];
    #pragma unroll
    for (int v = 0; v < 8; ++v) {
      const float mn = fmaxf(mrow[v], tm[v]);
      corr[v] = __expf(mrow[v] - mn);
      mrow[v] = mn;
      rs[v]   = 0.0f;
    }
    const int prow = (lane >> 4) * 8;
    #pragma unroll
    for (int tn = 0; tn < 4; ++tn)
      #pragma unroll
      for (int v = 0; v < 8; ++v) {
        const float p = __expf(sc[tn][v] - mrow[v]);
        rs[v] += p;
        pw[(prow + v) * LDK + tn * 16 + nr] = (bf16)p;   // D-layout -> LDS
      }
    #pragma unroll
    for (int off = 8; off >= 1; off >>= 1)
      #pragma unroll
      for (int v = 0; v < 8; ++v)
        rs[v] += __shfl_xor(rs[v], off, 32);
    #pragma unroll
    for (int v = 0; v < 8; ++v) lrow[v] = lrow[v] * corr[v] + rs[v];
    #pragma unroll
    for (int d = 0; d < 4; ++d)
      #pragma unroll
      for (int v = 0; v < 8; ++v) o[d][v] *= corr[v];

    __builtin_amdgcn_wave_barrier();           // DS ops are in-order per wave

    #pragma unroll
    for (int ks = 0; ks < 2; ++ks) {
      BF16Frag pf;
      const bf16* pp = &pw[nr * LDK + ks * 32 + ak];
      pf.h[0] = *(const v8bf*)pp;
      pf.h[1] = *(const v8bf*)(pp + 16);
      #pragma unroll
      for (int dt = 0; dt < 4; ++dt) {
        BF16Frag vf;
        const bf16* vp = &lv[(dt * 16 + nr) * LDK + ks * 32 + kb16];
        vf.h[0] = *(const v8bf*)vp;
        vf.h[1] = *(const v8bf*)(vp + 16);
        o[dt] = wmma_bf16(pf, vf, o[dt]);
      }
    }
  };

#ifdef USE_ASYNC_LDS
  // ---- async double-buffered K/V pipeline (4x b128 async copies / thread) ----
  auto issue_tile = [&](int kb, int buf) {
    const bf16* g  = Kh + ((size_t)bh * SEQ + kb + krow) * HEAD_DIM + kcol;
    const bf16* gv = Vt + ((size_t)bh * HEAD_DIM + krow) * SEQ + kb + kcol;
    bf16* lk = &ldsK[buf][krow * LDK + kcol];
    bf16* lv = &ldsV[buf][krow * LDK + kcol];
    __builtin_amdgcn_global_load_async_to_lds_b128((as1_v4i*)g,        (as3_v4i*)lk,       0, 0);
    __builtin_amdgcn_global_load_async_to_lds_b128((as1_v4i*)(g + 8),  (as3_v4i*)(lk + 8), 0, 0);
    __builtin_amdgcn_global_load_async_to_lds_b128((as1_v4i*)gv,       (as3_v4i*)lv,       0, 0);
    __builtin_amdgcn_global_load_async_to_lds_b128((as1_v4i*)(gv + 8), (as3_v4i*)(lv + 8), 0, 0);
  };

  issue_tile(0, 0);
  for (int it = 0; it < NIT; ++it) {
    const int kb  = it * 64;
    const int cur = it & 1;
    __syncthreads();                           // readers of buf cur^1 are done
    issue_tile((kb + 64) & (SEQ - 1), cur ^ 1);// last iter: harmless wrapped prefetch
    __builtin_amdgcn_s_wait_asynccnt(4);       // in-order: current tile's 4 ops done
    __syncthreads();                           // whole tile `cur` visible
    compute_tile(kb, ldsK[cur], ldsV[cur]);
  }
#else
  // ---- synchronous single-buffer fallback ----
  for (int it = 0; it < NIT; ++it) {
    const int kb = it * 64;
    __syncthreads();
    {
      const bf16* g = Kh + ((size_t)bh * SEQ + kb + krow) * HEAD_DIM + kcol;
      *(v8bf*)&ldsK[0][krow * LDK + kcol]     = *(const v8bf*)g;
      *(v8bf*)&ldsK[0][krow * LDK + kcol + 8] = *(const v8bf*)(g + 8);
      const bf16* gv = Vt + ((size_t)bh * HEAD_DIM + krow) * SEQ + kb + kcol;
      *(v8bf*)&ldsV[0][krow * LDK + kcol]     = *(const v8bf*)gv;
      *(v8bf*)&ldsV[0][krow * LDK + kcol + 8] = *(const v8bf*)(gv + 8);
    }
    __syncthreads();
    compute_tile(kb, ldsK[0], ldsV[0]);
  }
#endif

  // ---- normalize + write merged context [B, S, D_MODEL] ----
  #pragma unroll
  for (int v = 0; v < 8; ++v) {
    const float inv = (lrow[v] > 0.0f) ? 1.0f / lrow[v] : 0.0f;
    const int q = q0 + (lane >> 4) * 8 + v;
    bf16* op = out + ((size_t)b * SEQ + q) * D_MODEL + h * HEAD_DIM;
    #pragma unroll
    for (int dt = 0; dt < 4; ++dt)
      op[dt * 16 + (lane & 15)] = (bf16)(o[dt][v] * inv);
  }
}

// ---------------------------------------------------------------------------
extern "C" void kernel_launch(void* const* d_in, const int* in_sizes, int n_in,
                              void* d_out, int out_size, void* d_ws, size_t ws_size,
                              hipStream_t stream) {
  (void)in_sizes; (void)n_in; (void)out_size; (void)ws_size;

  const float* query = (const float*)d_in[0];
  const float* key   = (const float*)d_in[1];
  const float* value = (const float*)d_in[2];
  const unsigned char* mask = (const unsigned char*)d_in[3];
  const float* w_q = (const float*)d_in[4];
  const float* b_q = (const float*)d_in[5];
  const float* w_k = (const float*)d_in[6];
  const float* b_k = (const float*)d_in[7];
  const float* w_v = (const float*)d_in[8];
  const float* b_v = (const float*)d_in[9];
  const float* w_o = (const float*)d_in[10];
  const float* b_o = (const float*)d_in[11];

  // Workspace: 4 bf16 buffers of B*H*S*hd = 4Mi elements (8 MB each, 32 MB total)
  const size_t nHead = (size_t)BATCH * NUM_HEADS * SEQ * HEAD_DIM;
  bf16* Qh  = (bf16*)d_ws;
  bf16* Khd = Qh + nHead;
  bf16* Vt  = Khd + nHead;
  bf16* Att = Vt + nHead;

  dim3 gg(D_MODEL / 128, MTOK / 128);          // 8 x 32 blocks
  gemm_bias_kernel<0><<<gg, 256, 0, stream>>>(query, w_q, b_q, Qh);
  gemm_bias_kernel<0><<<gg, 256, 0, stream>>>(key,   w_k, b_k, Khd);
  gemm_bias_kernel<2><<<gg, 256, 0, stream>>>(value, w_v, b_v, Vt);

  dim3 ga(SEQ / 128, BATCH * NUM_HEADS);       // 16 x 32 blocks
  attention_kernel<<<ga, 256, 0, stream>>>(Qh, Khd, Vt, mask, Att);

  gemm_bias_kernel<1><<<gg, 256, 0, stream>>>(Att, w_o, b_o, (float*)d_out);
}
